// biLSTM_82429012345517
// MI455X (gfx1250) — matrix operand order repaired and checked by reading, not compiled, not run on hardware
//
#include <hip/hip_runtime.h>
#include <hip/hip_bf16.h>

#define BATCH 512
#define SEQT  1024
#define IDIM  64
#define HDIM  128
#define NCLS  10
#define NSTEP (SEQT - 1)
#define NGATE4H 512   // 4*HDIM gate columns

typedef __bf16 bf16_t;
typedef __attribute__((ext_vector_type(16))) __bf16 v16bf;
typedef __attribute__((ext_vector_type(8)))  __bf16 v8bf;
typedef __attribute__((ext_vector_type(4)))  __bf16 v4bf;
typedef __attribute__((ext_vector_type(8)))  float  v8f;
typedef __attribute__((ext_vector_type(4)))  float  f32x4;

// Load a 16-element bf16 fragment as two 16B LDS reads.
__device__ __forceinline__ v16bf ld_frag(const bf16_t* p0, const bf16_t* p1) {
  v8bf a = *(const v8bf*)p0;
  v8bf b = *(const v8bf*)p1;
  return __builtin_shufflevector(a, b, 0,1,2,3,4,5,6,7,8,9,10,11,12,13,14,15);
}

// Fast transcendentals on the trans pipe (v_exp_f32 / v_rcp_f32).
__device__ __forceinline__ float fast_exp(float x)  { return __builtin_amdgcn_exp2f(x * 1.4426950408889634f); }
__device__ __forceinline__ float fast_sigm(float x) { return __builtin_amdgcn_rcpf(1.0f + fast_exp(-x)); }
__device__ __forceinline__ float fast_tanh(float x) { return 1.0f - 2.0f * __builtin_amdgcn_rcpf(1.0f + fast_exp(2.0f * x)); }

// Dynamic-LDS layout (bytes); total 208,896 B <= 320 KB per WGP.
// WhT/WxT are only touched at setup (fragments then live in VGPRs).
#define OFF_WHT   0        // bf16 WhT [512][128] = 131072
#define OFF_WXT   131072   // bf16 WxT [512][64]  =  65536
#define OFF_XB    196608   // bf16 x   [2][16][64]  = 4096 (double buffered)
#define OFF_HB    200704   // bf16 h   [2][16][128] = 8192 (double buffered)
#define SMEM_BYTES 208896

extern "C" __global__ __launch_bounds__(256, 1)
void bilstm_tiles_kernel(const float* __restrict__ x,
                         const float* __restrict__ Wx_f, const float* __restrict__ Wh_f,
                         const float* __restrict__ b_f,
                         const float* __restrict__ Wx_b, const float* __restrict__ Wh_b,
                         const float* __restrict__ b_b,
                         const float* __restrict__ c0_f, const float* __restrict__ h0_f,
                         const float* __restrict__ c0_b, const float* __restrict__ h0_b,
                         float* __restrict__ hcat_out /* ws: [2][512][128] f32 */) {
  extern __shared__ char smem[];
  bf16_t* WhT = (bf16_t*)(smem + OFF_WHT);
  bf16_t* WxT = (bf16_t*)(smem + OFF_WXT);
  bf16_t* xb0 = (bf16_t*)(smem + OFF_XB);
  bf16_t* xb1 = xb0 + 16 * IDIM;
  bf16_t* hb0 = (bf16_t*)(smem + OFF_HB);
  bf16_t* hb1 = hb0 + 16 * HDIM;

  const int tid = threadIdx.x;
  const int dir = blockIdx.x >> 5;    // 0 = forward, 1 = backward
  const int bt  = blockIdx.x & 31;    // batch tile
  const int b0  = bt * 16;

  const float* Wx = dir ? Wx_b : Wx_f;
  const float* Wh = dir ? Wh_b : Wh_f;
  const float* bg = dir ? b_b  : b_f;
  const float* c0 = dir ? c0_b : c0_f;
  const float* h0 = dir ? h0_b : h0_f;

  // ---- one-time weight staging: bf16, transposed to [n][k] (B-fragment friendly) ----
  for (int idx = tid; idx < NGATE4H * HDIM; idx += 256) {
    int n = idx >> 7, k = idx & 127;
    WhT[idx] = (bf16_t)Wh[((n >> 7) << 14) + k * HDIM + (n & 127)];
  }
  for (int idx = tid; idx < NGATE4H * IDIM; idx += 256) {
    int n = idx >> 6, k = idx & 63;
    WxT[idx] = (bf16_t)Wx[((n >> 7) << 13) + k * HDIM + (n & 127)];
  }

  // ---- stage h0 -> hb0 (cooperative, row-major [16][128]) ----
  {
    int r = tid >> 4, jb = (tid & 15) << 3;
    v8bf h16;
    #pragma unroll
    for (int v = 0; v < 8; ++v) h16[v] = (bf16_t)h0[(b0 + r) * HDIM + jb + v];
    *(v8bf*)(hb0 + r * HDIM + jb) = h16;
  }

  // ---- stage x_0 -> xb0: each thread owns one float4 of the 16x64 tile ----
  const int rx = tid >> 4;
  const int kx = (tid & 15) << 2;
  const float* xrow = x + (long)(b0 + rx) * SEQT * IDIM + kx;
  {
    int t0 = dir ? (SEQT - 2) : 0;
    f32x4 x0 = *(const f32x4*)(xrow + (long)t0 * IDIM);
    v4bf xbv;
    #pragma unroll
    for (int v = 0; v < 4; ++v) xbv[v] = (bf16_t)x0[v];
    *(v4bf*)(xb0 + rx * IDIM + kx) = xbv;
  }
  __syncthreads();

  const int lane = tid & 31;
  const int wave = tid >> 5;
  const int m16  = lane & 15;
  const int hi   = (lane >> 4) & 1;
  const int hi8  = hi * 8;
  const int hi16 = hi * 16;
  const int j0   = wave * 16;        // this wave owns hidden slice [j0, j0+16)
  const int col  = j0 + m16;         // lane's hidden column

  // ---- per-lane recurrent state: (rows v+hi8, hidden col) of the 16x128 tile ----
  float creg[8], hcur[8];
  #pragma unroll
  for (int v = 0; v < 8; ++v) {
    int m = v + hi8;
    creg[v] = c0[(b0 + m) * HDIM + col];
    hcur[v] = h0[(b0 + m) * HDIM + col];
  }
  const float big = bg[col];
  const float bii = bg[HDIM + col];
  const float bif = bg[2 * HDIM + col];
  const float bio = bg[3 * HDIM + col];

  // ---- hoist ALL weight B-fragments into VGPRs (loop-invariant): 24 frags = 192 VGPRs ----
  v16bf wxf[2][4];   // [k-chunk][gate]
  v16bf whf[4][4];
  #pragma unroll
  for (int kc = 0; kc < 2; ++kc)
    #pragma unroll
    for (int g = 0; g < 4; ++g) {
      const bf16_t* p = WxT + (g * HDIM + col) * IDIM + kc * 32 + hi8;
      wxf[kc][g] = ld_frag(p, p + 16);
    }
  #pragma unroll
  for (int kc = 0; kc < 4; ++kc)
    #pragma unroll
    for (int g = 0; g < 4; ++g) {
      const bf16_t* p = WhT + (g * HDIM + col) * HDIM + kc * 32 + hi8;
      whf[kc][g] = ld_frag(p, p + 16);
    }

  #pragma unroll 1
  for (int s = 0; s < NSTEP; ++s) {
    const int par = s & 1;
    const bf16_t* xr_buf = par ? xb1 : xb0;
    bf16_t*       xw_buf = par ? xb0 : xb1;
    const bf16_t* hr_buf = par ? hb1 : hb0;
    bf16_t*       hw_buf = par ? hb0 : hb1;

    // prefetch next x tile into regs; latency hidden under WMMA phase
    f32x4 xnext = {0.f, 0.f, 0.f, 0.f};
    const bool more = (s + 1) < NSTEP;
    if (more) {
      int tn = dir ? (SEQT - 2 - (s + 1)) : (s + 1);
      xnext = *(const f32x4*)(xrow + (long)tn * IDIM);
    }

    // ---- A fragments (ISA 16-bit A layout) ----
    v16bf xa[2], ha[4];
    #pragma unroll
    for (int kc = 0; kc < 2; ++kc) {
      const bf16_t* q = xr_buf + m16 * IDIM + kc * 32 + hi8;
      xa[kc] = ld_frag(q, q + 16);
    }
    #pragma unroll
    for (int kc = 0; kc < 4; ++kc) {
      const bf16_t* q = hr_buf + m16 * HDIM + kc * 32 + hi8;
      ha[kc] = ld_frag(q, q + 16);
    }

    // ---- 4 independent gate-tile chains, all operands in registers: 24 WMMAs ----
    v8f acc[4];
    #pragma unroll
    for (int i = 0; i < 4; ++i)
      #pragma unroll
      for (int j = 0; j < 8; ++j) acc[i][j] = 0.0f;

    #pragma unroll
    for (int kc = 0; kc < 2; ++kc)
      #pragma unroll
      for (int g = 0; g < 4; ++g)
        acc[g] = __builtin_amdgcn_wmma_f32_16x16x32_bf16(
            false, xa[kc], false, wxf[kc][g], (short)0, acc[g], false, false);
    #pragma unroll
    for (int kc = 0; kc < 4; ++kc)
      #pragma unroll
      for (int g = 0; g < 4; ++g)
        acc[g] = __builtin_amdgcn_wmma_f32_16x16x32_bf16(
            false, ha[kc], false, whf[kc][g], (short)0, acc[g], false, false);

    // ---- cell/hidden update entirely in registers (lane owns g,i,f,o of its elems) ----
    #pragma unroll
    for (int v = 0; v < 8; ++v) {
      float pg = acc[0][v] + big;
      float pi = acc[1][v] + bii;
      float pf = acc[2][v] + bif;
      float po = acc[3][v] + bio;
      creg[v] = fast_tanh(pg) * fast_sigm(pi) + creg[v] * fast_sigm(pf);
      hcur[v] = fast_tanh(creg[v]) * fast_sigm(po);
      hw_buf[(v + hi8) * HDIM + col] = (bf16_t)hcur[v];
    }

    // commit prefetched x tile to the other buffer
    if (more) {
      v4bf xbv;
      #pragma unroll
      for (int v = 0; v < 4; ++v) xbv[v] = (bf16_t)xnext[v];
      *(v4bf*)(xw_buf + rx * IDIM + kx) = xbv;
    }
    __syncthreads();   // single barrier per step (double-buffered staging)
  }

  // ---- final hidden state for this (dir, batch-tile) ----
  float* hout = hcat_out + (long)dir * BATCH * HDIM;
  #pragma unroll
  for (int v = 0; v < 8; ++v)
    hout[(long)(b0 + v + hi8) * HDIM + col] = hcur[v];
}

extern "C" __global__ __launch_bounds__(128)
void proj_softmax_kernel(const float* __restrict__ hcat, const float* __restrict__ Wph,
                         const float* __restrict__ bp, float* __restrict__ out) {
  int b = blockIdx.x * 128 + threadIdx.x;
  if (b >= BATCH) return;
  float lg[NCLS];
  #pragma unroll
  for (int c = 0; c < NCLS; ++c) lg[c] = bp[c];
  const float* hf = hcat + (long)b * HDIM;
  const float* hb = hcat + (long)BATCH * HDIM + (long)b * HDIM;
  for (int k = 0; k < HDIM; ++k) {
    float v = hf[k];
    #pragma unroll
    for (int c = 0; c < NCLS; ++c) lg[c] += v * Wph[k * NCLS + c];
  }
  for (int k = 0; k < HDIM; ++k) {
    float v = hb[k];
    #pragma unroll
    for (int c = 0; c < NCLS; ++c) lg[c] += v * Wph[(HDIM + k) * NCLS + c];
  }
  float mx = lg[0];
  #pragma unroll
  for (int c = 1; c < NCLS; ++c) mx = fmaxf(mx, lg[c]);
  float sum = 0.f;
  #pragma unroll
  for (int c = 0; c < NCLS; ++c) { lg[c] = __expf(lg[c] - mx); sum += lg[c]; }
  float inv = 1.0f / sum;
  #pragma unroll
  for (int c = 0; c < NCLS; ++c) out[b * NCLS + c] = lg[c] * inv;
}

extern "C" void kernel_launch(void* const* d_in, const int* in_sizes, int n_in,
                              void* d_out, int out_size, void* d_ws, size_t ws_size,
                              hipStream_t stream) {
  (void)in_sizes; (void)n_in; (void)out_size; (void)ws_size;
  const float* x    = (const float*)d_in[0];
  const float* Wx_f = (const float*)d_in[1];
  const float* Wh_f = (const float*)d_in[2];
  const float* b_f  = (const float*)d_in[3];
  const float* Wx_b = (const float*)d_in[4];
  const float* Wh_b = (const float*)d_in[5];
  const float* b_b  = (const float*)d_in[6];
  const float* Wph  = (const float*)d_in[7];
  const float* bp   = (const float*)d_in[8];
  const float* c0_f = (const float*)d_in[9];
  const float* h0_f = (const float*)d_in[10];
  const float* c0_b = (const float*)d_in[11];
  const float* h0_b = (const float*)d_in[12];
  float* hcat = (float*)d_ws;   // [2][512][128] f32

  hipFuncSetAttribute((const void*)bilstm_tiles_kernel,
                      hipFuncAttributeMaxDynamicSharedMemorySize, SMEM_BYTES);

  // 64 workgroups = 2 directions x 32 batch tiles; 8 wave32 waves each
  bilstm_tiles_kernel<<<dim3(64), dim3(256), SMEM_BYTES, stream>>>(
      x, Wx_f, Wh_f, b_f, Wx_b, Wh_b, b_b, c0_f, h0_f, c0_b, h0_b, hcat);

  proj_softmax_kernel<<<dim3(4), dim3(128), 0, stream>>>(hcat, Wph, bp, (float*)d_out);
}
